// JAttention_9698036155149
// MI455X (gfx1250) — matrix operand order repaired and checked by reading, compiled, not run on hardware
//
#include <hip/hip_runtime.h>
#include <hip/hip_bf16.h>

// ---------------------------------------------------------------------------
// CDNA5 (gfx1250) attention block: QKV proj + RoPE + causal flash-attn + O proj
// Matmuls via v_wmma_f32_16x16x32_bf16 (wave32). Tile staging via the Tensor
// Data Mover (tensor_load_to_lds + s_wait_tensorcnt) where available.
// ---------------------------------------------------------------------------

typedef __bf16 bf16_t;
typedef __bf16 v16bf __attribute__((ext_vector_type(16)));
typedef __bf16 v8bf  __attribute__((ext_vector_type(8)));
typedef float  v8f   __attribute__((ext_vector_type(8)));
typedef unsigned int v4u __attribute__((ext_vector_type(4)));
typedef int v8i __attribute__((ext_vector_type(8)));
typedef int v4i __attribute__((ext_vector_type(4)));

#define S_LEN   4096
#define HID     2048
#define NQH     16
#define NKVH    4
#define HD      128

#if __has_builtin(__builtin_amdgcn_tensor_load_to_lds) && \
    __has_builtin(__builtin_amdgcn_s_wait_tensorcnt)
#define USE_TDM 1
#else
#define USE_TDM 0
#endif

__device__ inline v16bf make_frag(v8bf lo, v8bf hi) {
  union { v16bf v; v8bf h[2]; } u;
  u.h[0] = lo; u.h[1] = hi;
  return u.v;
}

__device__ inline v8f zero8() {
  v8f z;
#pragma unroll
  for (int i = 0; i < 8; i++) z[i] = 0.0f;
  return z;
}

__device__ inline v8f wmma_bf16(v16bf a, v16bf b, v8f c) {
  // D = A(16x32 bf16) * B(32x16 bf16) + C(16x16 f32)
  return __builtin_amdgcn_wmma_f32_16x16x32_bf16(false, a, false, b,
                                                 (short)0, c, false, false);
}

#if USE_TDM
// ---------------------------------------------------------------------------
// Tensor Data Mover: 2D bf16 tile load global->LDS with LDS row padding.
// D# bitfields per CDNA5 ISA ch.8 (group0 128b, group1 256b; groups 2/3 zero
// for 2D tiles). lds_off = wave-relative LDS byte address (low 32 bits of the
// generic pointer, per the LDS aperture mapping). data_size=1 (2 bytes).
//   tile_d0 = contiguous elems per row, tile_d1 = rows,
//   stride  = global row stride in elems,
//   pad_int_code: LDS DWORDs between pads = 2<<code; pad_amt_code: pad DWORDs-1
// This toolchain exposes the 6-arg builtin:
//   (uint32x4 g0, int32x8 g1, int32x4, int32x4, int32x8, i32 cpol)
// ---------------------------------------------------------------------------
__device__ inline void tdm_load_2d_bf16(unsigned lds_off, const void* gptr,
                                        unsigned tile_d0, unsigned tile_d1,
                                        unsigned long stride_elems,
                                        unsigned pad_int_code,
                                        unsigned pad_amt_code) {
  unsigned long ga = (unsigned long)gptr;
  v4u g0;
  g0[0] = 1u;                                       // count=1 (valid), user mode
  g0[1] = lds_off;                                  // lds_addr (bytes)
  g0[2] = (unsigned)(ga & 0xffffffffu);             // global_addr[31:0]
  g0[3] = (unsigned)((ga >> 32) & 0x01ffffffu)      // global_addr[56:32]
        | (2u << 30);                               // type=2 ("image")
  v8i g1;
  g1[0] = (int)((1u << 16)                          // data_size=1 -> 2 bytes
              | (1u << 20)                          // pad_enable
              | (pad_int_code << 22)                // pad_interval
              | (pad_amt_code << 25));              // pad_amount
  g1[1] = (int)(tile_d0 << 16);                     // tensor_dim0[15:0] (=tile)
  g1[2] = (int)((tile_d0 >> 16) | (tile_d1 << 16)); // dim0 hi | tensor_dim1 lo
  g1[3] = (int)((tile_d1 >> 16) | (tile_d0 << 16)); // dim1 hi | tile_dim0
  g1[4] = (int)(tile_d1 & 0xffffu);                 // tile_dim1 ; tile_dim2=0
  g1[5] = (int)(stride_elems & 0xffffffffu);        // tensor_dim0_stride lo
  g1[6] = (int)((stride_elems >> 32) & 0xffffu);    // stride hi ; dim1_stride=0
  g1[7] = 0;
  v4i z4 = {0, 0, 0, 0};
  v8i z8 = {0, 0, 0, 0, 0, 0, 0, 0};
  __builtin_amdgcn_tensor_load_to_lds(g0, g1, z4, z4, z8, 0);
}

__device__ inline unsigned lds_off_of(const void* p) {
  return (unsigned)(unsigned long)p;   // generic shared ptr low 32b = LDS offset
}
#endif

// ---------------------------------------------------------------------------
// f32 -> bf16 elementwise convert
// ---------------------------------------------------------------------------
__global__ void cvt_bf16_kernel(const float* __restrict__ in,
                                bf16_t* __restrict__ out, int n) {
  int i = blockIdx.x * 256 + threadIdx.x;
  if (i < n) out[i] = (bf16_t)in[i];
}

// ---------------------------------------------------------------------------
// W f32 [K][N] -> Wt bf16 [N][K]
// ---------------------------------------------------------------------------
__global__ void transpose_cvt_kernel(const float* __restrict__ W,
                                     bf16_t* __restrict__ Wt, int K, int N) {
  int i = blockIdx.x * 256 + threadIdx.x;
  if (i < N * K) {
    int n = i / K, k = i % K;
    Wt[i] = (bf16_t)W[(long)k * N + n];
  }
}

// ---------------------------------------------------------------------------
// In-place RoPE on bf16 activations [S][nheads*128]; pos = 8192 + s.
// ---------------------------------------------------------------------------
__global__ void rope_kernel(bf16_t* __restrict__ X, int nheads) {
  int i = blockIdx.x * 256 + threadIdx.x;
  int total = S_LEN * nheads * 64;
  if (i >= total) return;
  int d = i % 64;
  int t = i / 64;
  int h = t % nheads;
  int s = t / nheads;
  float inv = __expf((float)d * -0.14391156831212787f); // -ln(10000)/64
  float ang = (8192.0f + (float)s) * inv;
  float cv = cosf(ang), sv = sinf(ang);
  long base = ((long)s * nheads + h) * HD + d;
  float x1 = (float)X[base];
  float x2 = (float)X[base + 64];
  X[base]      = (bf16_t)(x1 * cv - x2 * sv);
  X[base + 64] = (bf16_t)(x2 * cv + x1 * sv);
}

// ---------------------------------------------------------------------------
// bf16 GEMM: C[M][N] = A[M][K] @ B[K][N], B given transposed (Bt[N][K]).
// Block 64x128, 8 waves, wave tile 32x32, K-step 32. Templated output type.
// ---------------------------------------------------------------------------
template <bool OUT_BF16>
__global__ __launch_bounds__(256) void gemm_bf16_kernel(
    const bf16_t* __restrict__ A, const bf16_t* __restrict__ Bt,
    void* __restrict__ Cout, int M, int N, int K) {
  __shared__ bf16_t As[64 * 40];   // 64 x 32, row stride 40 elems (pad 8)
  __shared__ bf16_t Bs[128 * 40];  // 128 x 32

  const int tid  = threadIdx.x;
  const int w    = tid >> 5;
  const int lane = tid & 31;
  const int lm   = lane & 15;
  const int hh   = lane >> 4;
  const int wm   = w & 1;
  const int wn   = w >> 1;
  const int m0   = blockIdx.y * 64;
  const int n0   = blockIdx.x * 128;

  v8f acc[2][2];
#pragma unroll
  for (int i = 0; i < 2; i++)
#pragma unroll
    for (int j = 0; j < 2; j++) acc[i][j] = zero8();

  for (int k0 = 0; k0 < K; k0 += 32) {
    __syncthreads();  // prior iteration's fragment reads complete
#if USE_TDM
    if (w == 0) {
      // 32-elem rows (64B = 16 DW -> interval code 3), pad 8 elems (4 DW -> 3)
      tdm_load_2d_bf16(lds_off_of(&As[0]), &A[(long)m0 * K + k0],
                       32u, 64u, (unsigned long)K, 3u, 3u);
      tdm_load_2d_bf16(lds_off_of(&Bs[0]), &Bt[(long)n0 * K + k0],
                       32u, 128u, (unsigned long)K, 3u, 3u);
      __builtin_amdgcn_s_wait_tensorcnt(0);
    }
#else
    {   // A tile: one 16B chunk per thread
      int r = tid >> 2, c8 = (tid & 3) * 8;
      *(uint4*)&As[r * 40 + c8] =
          *(const uint4*)&A[(long)(m0 + r) * K + k0 + c8];
    }
#pragma unroll
    for (int i = 0; i < 2; i++) {  // Bt tile: two chunks per thread
      int idx = tid + i * 256;
      int r = idx >> 2, c8 = (idx & 3) * 8;
      *(uint4*)&Bs[r * 40 + c8] =
          *(const uint4*)&Bt[(long)(n0 + r) * K + k0 + c8];
    }
#endif
    __syncthreads();

    v16bf af[2], bfr[2];
#pragma unroll
    for (int fm = 0; fm < 2; fm++) {
      int row = wm * 32 + fm * 16 + lm;
      af[fm] = make_frag(*(const v8bf*)&As[row * 40 + hh * 8],
                         *(const v8bf*)&As[row * 40 + hh * 8 + 16]);
    }
#pragma unroll
    for (int fn = 0; fn < 2; fn++) {
      int row = wn * 32 + fn * 16 + lm;
      bfr[fn] = make_frag(*(const v8bf*)&Bs[row * 40 + hh * 8],
                          *(const v8bf*)&Bs[row * 40 + hh * 8 + 16]);
    }
#pragma unroll
    for (int fm = 0; fm < 2; fm++)
#pragma unroll
      for (int fn = 0; fn < 2; fn++)
        acc[fm][fn] = wmma_bf16(af[fm], bfr[fn], acc[fm][fn]);
  }

  // Epilogue. C layout: VGPR r -> row r + 8*(lane/16), col lane%16.
#pragma unroll
  for (int fm = 0; fm < 2; fm++)
#pragma unroll
    for (int fn = 0; fn < 2; fn++)
#pragma unroll
      for (int r = 0; r < 8; r++) {
        int row = m0 + wm * 32 + fm * 16 + r + 8 * hh;
        int col = n0 + wn * 32 + fn * 16 + lm;
        float v = acc[fm][fn][r];
        if constexpr (OUT_BF16)
          ((bf16_t*)Cout)[(long)row * N + col] = (bf16_t)v;
        else
          ((float*)Cout)[(long)row * N + col] = v;
      }
}

// ---------------------------------------------------------------------------
// Causal flash attention. Grid (S/64, 16 heads), 128 threads (4 waves).
// ---------------------------------------------------------------------------
__global__ __launch_bounds__(128) void attn_kernel(
    const bf16_t* __restrict__ Q, const bf16_t* __restrict__ Kb,
    const bf16_t* __restrict__ Vb, bf16_t* __restrict__ O) {
  __shared__ bf16_t Qs[64 * 136];   // 64 x 128, stride 136 (pad 8)
  __shared__ bf16_t Ks[64 * 136];
  __shared__ bf16_t Vt[128 * 72];   // V transposed: [d][key], stride 72
  __shared__ bf16_t Ps[4 * 16 * 72];

  const int qb   = blockIdx.x;
  const int head = blockIdx.y;
  const int q0   = qb * 64;
  const int kvh  = head >> 2;       // KV_GROUPS = 4
  const int tid  = threadIdx.x;
  const int w    = tid >> 5;
  const int lane = tid & 31;
  const int lm   = lane & 15;
  const int hh   = lane >> 4;

  // ---- Q tile (64 x 128) ----
#if USE_TDM
  if (w == 0) {
    // 128-elem rows (256B = 64 DW -> interval code 5), pad 8 elems (4 DW -> 3)
    tdm_load_2d_bf16(lds_off_of(&Qs[0]),
                     &Q[(long)q0 * (NQH * HD) + head * HD],
                     128u, 64u, (unsigned long)(NQH * HD), 5u, 3u);
    __builtin_amdgcn_s_wait_tensorcnt(0);
  }
#else
#pragma unroll
  for (int i = 0; i < 8; i++) {
    int c = tid + i * 128;
    int r = c >> 4, off = (c & 15) * 8;
    *(uint4*)&Qs[r * 136 + off] =
        *(const uint4*)&Q[(long)(q0 + r) * (NQH * HD) + head * HD + off];
  }
#endif

  v8f oacc[8];
#pragma unroll
  for (int i = 0; i < 8; i++) oacc[i] = zero8();
  float mrow[8], lrow[8];
#pragma unroll
  for (int r = 0; r < 8; r++) { mrow[r] = -3.0e38f; lrow[r] = 0.0f; }

  const float scale = 0.08838834764831845f;  // 1/sqrt(128)

  for (int kb = 0; kb <= qb; kb++) {
    const int k0 = kb * 64;
    __syncthreads();  // previous iteration's K/V fragment reads complete
#if USE_TDM
    if (w == 0) {
      tdm_load_2d_bf16(lds_off_of(&Ks[0]),
                       &Kb[(long)k0 * (NKVH * HD) + kvh * HD],
                       128u, 64u, (unsigned long)(NKVH * HD), 5u, 3u);
      __builtin_amdgcn_s_wait_tensorcnt(0);
    }
#else
#pragma unroll
    for (int i = 0; i < 8; i++) {  // K tile 64x128
      int c = tid + i * 128;
      int r = c >> 4, off = (c & 15) * 8;
      *(uint4*)&Ks[r * 136 + off] =
          *(const uint4*)&Kb[(long)(k0 + r) * (NKVH * HD) + kvh * HD + off];
    }
#endif
#pragma unroll
    for (int i = 0; i < 8; i++) {  // V tile, transposed into LDS
      int c = tid + i * 128;
      int r = c >> 4, off = (c & 15) * 8;
      uint4 raw = *(const uint4*)&Vb[(long)(k0 + r) * (NKVH * HD) + kvh * HD + off];
      const bf16_t* e = (const bf16_t*)&raw;
#pragma unroll
      for (int j = 0; j < 8; j++) Vt[(off + j) * 72 + r] = e[j];
    }
    __syncthreads();

    // ---- Scores S = Q K^T : wave rows w*16.., all 64 keys ----
    v8f sacc[4];
#pragma unroll
    for (int fn = 0; fn < 4; fn++) sacc[fn] = zero8();
#pragma unroll
    for (int d0 = 0; d0 < 128; d0 += 32) {
      int arow = w * 16 + lm;
      v16bf a = make_frag(*(const v8bf*)&Qs[arow * 136 + d0 + hh * 8],
                          *(const v8bf*)&Qs[arow * 136 + d0 + hh * 8 + 16]);
#pragma unroll
      for (int fn = 0; fn < 4; fn++) {
        int brow = fn * 16 + lm;
        v16bf b = make_frag(*(const v8bf*)&Ks[brow * 136 + d0 + hh * 8],
                            *(const v8bf*)&Ks[brow * 136 + d0 + hh * 8 + 16]);
        sacc[fn] = wmma_bf16(a, b, sacc[fn]);
      }
    }

    // ---- Scale + causal mask (diagonal block only) ----
#pragma unroll
    for (int fn = 0; fn < 4; fn++)
#pragma unroll
      for (int r = 0; r < 8; r++) {
        float v = sacc[fn][r] * scale;
        if (kb == qb) {
          int qg = q0 + w * 16 + r + 8 * hh;
          int kg = k0 + fn * 16 + lm;
          if (kg > qg) v = -3.0e38f;
        }
        sacc[fn][r] = v;
      }

    // ---- Row max (butterfly across 16-lane half) ----
    float pm[8];
#pragma unroll
    for (int r = 0; r < 8; r++) {
      float m = sacc[0][r];
#pragma unroll
      for (int fn = 1; fn < 4; fn++) m = fmaxf(m, sacc[fn][r]);
      pm[r] = m;
    }
#pragma unroll
    for (int off = 1; off < 16; off <<= 1)
#pragma unroll
      for (int r = 0; r < 8; r++)
        pm[r] = fmaxf(pm[r], __shfl_xor(pm[r], off, 32));

    // ---- Online softmax ----
    float alpha[8];
#pragma unroll
    for (int r = 0; r < 8; r++) {
      float mn = fmaxf(mrow[r], pm[r]);
      alpha[r] = __expf(mrow[r] - mn);
      mrow[r] = mn;
      lrow[r] *= alpha[r];
    }
#pragma unroll
    for (int nd = 0; nd < 8; nd++)
#pragma unroll
      for (int r = 0; r < 8; r++) oacc[nd][r] *= alpha[r];

    float psum[8];
#pragma unroll
    for (int r = 0; r < 8; r++) psum[r] = 0.0f;
#pragma unroll
    for (int fn = 0; fn < 4; fn++)
#pragma unroll
      for (int r = 0; r < 8; r++) {
        float p = __expf(sacc[fn][r] - mrow[r]);
        sacc[fn][r] = p;
        psum[r] += p;
      }
#pragma unroll
    for (int off = 1; off < 16; off <<= 1)
#pragma unroll
      for (int r = 0; r < 8; r++) psum[r] += __shfl_xor(psum[r], off, 32);
#pragma unroll
    for (int r = 0; r < 8; r++) lrow[r] += psum[r];

    // ---- P (C layout) -> LDS -> A layout ----
    bf16_t* Pw = &Ps[w * 16 * 72];
#pragma unroll
    for (int fn = 0; fn < 4; fn++)
#pragma unroll
      for (int r = 0; r < 8; r++)
        Pw[(r + 8 * hh) * 72 + fn * 16 + lm] = (bf16_t)sacc[fn][r];

    // ---- O += P @ V ----
#pragma unroll
    for (int kp = 0; kp < 2; kp++) {
      v16bf ap = make_frag(*(const v8bf*)&Pw[lm * 72 + kp * 32 + hh * 8],
                           *(const v8bf*)&Pw[lm * 72 + kp * 32 + hh * 8 + 16]);
#pragma unroll
      for (int nd = 0; nd < 8; nd++) {
        int vrow = nd * 16 + lm;
        v16bf bv = make_frag(*(const v8bf*)&Vt[vrow * 72 + kp * 32 + hh * 8],
                             *(const v8bf*)&Vt[vrow * 72 + kp * 32 + hh * 8 + 16]);
        oacc[nd] = wmma_bf16(ap, bv, oacc[nd]);
      }
    }
  }

  // ---- Normalize + store bf16 attention output [S][NQH*HD] ----
  float inv[8];
#pragma unroll
  for (int r = 0; r < 8; r++) inv[r] = 1.0f / lrow[r];
#pragma unroll
  for (int nd = 0; nd < 8; nd++)
#pragma unroll
    for (int r = 0; r < 8; r++) {
      int row = q0 + w * 16 + r + 8 * hh;
      int col = head * HD + nd * 16 + lm;
      O[(long)row * (NQH * HD) + col] = (bf16_t)(oacc[nd][r] * inv[r]);
    }
}

// ---------------------------------------------------------------------------
extern "C" void kernel_launch(void* const* d_in, const int* in_sizes, int n_in,
                              void* d_out, int out_size, void* d_ws, size_t ws_size,
                              hipStream_t stream) {
  const float* x  = (const float*)d_in[0];
  const float* wq = (const float*)d_in[1];
  const float* wk = (const float*)d_in[2];
  const float* wv = (const float*)d_in[3];
  const float* wo = (const float*)d_in[4];
  float* out = (float*)d_out;

  char* ws = (char*)d_ws;
  bf16_t* Xbf = (bf16_t*)(ws + 0);                 // 16MB (reused as AttnOut)
  bf16_t* Wt  = (bf16_t*)(ws + (16L << 20));       // 8MB  (per-GEMM transposed W)
  bf16_t* Qb  = (bf16_t*)(ws + (24L << 20));       // 16MB
  bf16_t* Kb  = (bf16_t*)(ws + (40L << 20));       // 4MB
  bf16_t* Vb  = (bf16_t*)(ws + (44L << 20));       // 4MB
  bf16_t* Ao  = Xbf;                               // attention output, bf16

  const int nx = S_LEN * HID;
  cvt_bf16_kernel<<<(nx + 255) / 256, 256, 0, stream>>>(x, Xbf, nx);

  // ---- Q projection ----
  transpose_cvt_kernel<<<(HID * (NQH * HD) + 255) / 256, 256, 0, stream>>>(
      wq, Wt, HID, NQH * HD);
  gemm_bf16_kernel<true><<<dim3((NQH * HD) / 128, S_LEN / 64), 256, 0, stream>>>(
      Xbf, Wt, Qb, S_LEN, NQH * HD, HID);

  // ---- K projection ----
  transpose_cvt_kernel<<<(HID * (NKVH * HD) + 255) / 256, 256, 0, stream>>>(
      wk, Wt, HID, NKVH * HD);
  gemm_bf16_kernel<true><<<dim3((NKVH * HD) / 128, S_LEN / 64), 256, 0, stream>>>(
      Xbf, Wt, Kb, S_LEN, NKVH * HD, HID);

  // ---- V projection ----
  transpose_cvt_kernel<<<(HID * (NKVH * HD) + 255) / 256, 256, 0, stream>>>(
      wv, Wt, HID, NKVH * HD);
  gemm_bf16_kernel<true><<<dim3((NKVH * HD) / 128, S_LEN / 64), 256, 0, stream>>>(
      Xbf, Wt, Vb, S_LEN, NKVH * HD, HID);

  // ---- RoPE (in place) ----
  rope_kernel<<<(S_LEN * NQH * 64) / 256, 256, 0, stream>>>(Qb, NQH);
  rope_kernel<<<(S_LEN * NKVH * 64) / 256, 256, 0, stream>>>(Kb, NKVH);

  // ---- causal flash attention ----
  attn_kernel<<<dim3(S_LEN / 64, NQH), 128, 0, stream>>>(Qb, Kb, Vb, Ao);

  // ---- output projection -> f32 d_out ----
  transpose_cvt_kernel<<<((NQH * HD) * HID + 255) / 256, 256, 0, stream>>>(
      wo, Wt, NQH * HD, HID);
  gemm_bf16_kernel<false><<<dim3(HID / 128, S_LEN / 64), 256, 0, stream>>>(
      Ao, Wt, out, S_LEN, HID, NQH * HD);
}
